// RecurrentGenerator_7894149890299
// MI455X (gfx1250) — matrix operand order repaired and checked by reading, not compiled, not run on hardware
//
#include <hip/hip_runtime.h>

// Problem constants (match reference).
#define BB 8192
#define HH 512
#define EE 512
#define VV 2048
#define TT 24
#define G3 1536   // 3*HH

typedef __attribute__((ext_vector_type(16))) __bf16 v16bf;
typedef __attribute__((ext_vector_type(8)))  float  v8f;

// ---------------- wave32 reduction helpers ----------------
__device__ __forceinline__ float wave_sum(float v) {
#pragma unroll
  for (int o = 16; o > 0; o >>= 1) v += __shfl_down(v, o, 32);
  return v;
}
__device__ __forceinline__ float wave_max(float v) {
#pragma unroll
  for (int o = 16; o > 0; o >>= 1) v = fmaxf(v, __shfl_down(v, o, 32));
  return v;
}

__device__ __forceinline__ float sigmoidf_fast(float x) {
  return 1.0f / (1.0f + __expf(-x));
}

// splitmix64 -> uniform in (0,1)
__device__ __forceinline__ float u01(unsigned long long x) {
  x += 0x9E3779B97F4A7C15ull;
  x = (x ^ (x >> 30)) * 0xBF58476D1CE4E5B9ull;
  x = (x ^ (x >> 27)) * 0x94D049BB133111EBull;
  x ^= x >> 31;
  return ((float)(x >> 40) + 0.5f) * (1.0f / 16777216.0f);
}

// ---------------- conversion / init kernels ----------------
__global__ void k_f32_to_bf16(const float* __restrict__ src,
                              __bf16* __restrict__ dst, int n) {
  int i = blockIdx.x * blockDim.x + threadIdx.x;
  if (i < n) dst[i] = (__bf16)src[i];
}

__global__ void k_init_h(const float* __restrict__ emb, float* __restrict__ h,
                         __bf16* __restrict__ hb, int n) {
  int i = blockIdx.x * blockDim.x + threadIdx.x;
  if (i < n) { float v = emb[i]; h[i] = v; hb[i] = (__bf16)v; }
}

__global__ void k_init_x(const float* __restrict__ sos, __bf16* __restrict__ xb, int n) {
  int i = blockIdx.x * blockDim.x + threadIdx.x;
  if (i < n) xb[i] = (__bf16)sos[i & (EE - 1)];
}

// ---------------- WMMA bf16 GEMM with LDS-staged B ----------------
// C[M,N] = A[M,K] @ W[N,K]^T + bias[N]    (both A and W are K-contiguous)
//
// Block = 256 threads = 8 waves. Block tile: 128 M x 128 N.
// Wave w owns rows [m0, m0+16), all 128 cols via 8 f32 16x16 accumulators.
// Per K=32 chunk: B tile (128x32 bf16 = 8KB) is staged into LDS in WMMA
// *fragment layout* -- lane L of j-tile j needs exactly 32 contiguous bytes
// of row-major W, so the cooperative load is 256 threads x 32B, fully
// coalesced, and shared by all 8 waves (8x less L2 traffic than private
// loads). Compute side: 1 A fragment (global b128 loads) feeds 8 independent
// WMMAs whose B fragments are single aligned 32B LDS reads (ds_load_b128).
// Lane mapping: A and B use contiguous k = 16*(lane/16) + s; f32 C 16x16 per CDNA5 ISA 7.12.2.
__global__ void __launch_bounds__(256)
k_gemm_bf16(const __bf16* __restrict__ A, const __bf16* __restrict__ W,
            const float* __restrict__ bias, float* __restrict__ C,
            int N, int K) {
  const int tid  = threadIdx.x;
  const int lane = tid & 31;
  const int wv   = tid >> 5;
  const int m0 = (blockIdx.x * 8 + wv) * 16;
  const int n0 = blockIdx.y * 128;

  // B staging buffer: [j(8)][lane(32)][16 bf16] = 8KB, 32B-aligned chunks
  __shared__ __attribute__((aligned(32))) __bf16 lb[8 * 32 * 16];

  // cooperative B loader: thread -> (j = wv, frag-lane = lane), 32B each
  const int brow  = n0 + 16 * wv + (lane & 15);
  const int bkoff = (lane & 16) ? 16 : 0;
  const __bf16* bsrc = W + (size_t)brow * K + bkoff;
  uint4* ldst = (uint4*)&lb[(wv * 32 + lane) * 16];

  // A fragment source (private rows)
  const __bf16* arow = A + (size_t)(m0 + (lane & 15)) * K;
  const int akb = (lane & 16) ? 16 : 0;

  const v16bf* lbv = (const v16bf*)lb;

  v8f acc[8] = {v8f{}, v8f{}, v8f{}, v8f{}, v8f{}, v8f{}, v8f{}, v8f{}};

  for (int k0 = 0; k0 < K; k0 += 32) {
    // ---- stage B chunk into LDS ----
    __syncthreads();                       // prior chunk's reads complete
    {
      const uint4* s = (const uint4*)(bsrc + k0);
      uint4 q0 = s[0];
      uint4 q1 = s[1];
      ldst[0] = q0;
      ldst[1] = q1;
    }
    __builtin_prefetch(arow + k0 + 64, 0, 1);   // global_prefetch_b8: A chunk two ahead
    __builtin_prefetch(bsrc + k0 + 32, 0, 1);   // and next B chunk
    __syncthreads();                       // stores visible to all waves

    // ---- A fragment from global ----
    v16bf a;
#pragma unroll
    for (int v = 0; v < 8; ++v) {
      int k = k0 + akb + v * 2;
      a[2 * v]     = arow[k];
      a[2 * v + 1] = arow[k + 1];
    }

    // ---- 8 independent WMMAs, B fragments from LDS ----
#pragma unroll
    for (int j = 0; j < 8; ++j) {
      v16bf bmat = lbv[j * 32 + lane];
      acc[j] = __builtin_amdgcn_wmma_f32_16x16x32_bf16(
          false, a, false, bmat, (short)0, acc[j], false, false);
    }
  }

  // ---- epilogue: bias + store (f32 C layout: VGPR i -> row i + 8*(lane>=16)) ----
  const int col = lane & 15;
  const int rb  = (lane & 16) ? 8 : 0;
#pragma unroll
  for (int j = 0; j < 8; ++j) {
    int n = n0 + 16 * j + col;
    float bv = bias[n];
#pragma unroll
    for (int i = 0; i < 8; ++i)
      C[(size_t)(m0 + rb + i) * N + n] = acc[j][i] + bv;
  }
}

// ---------------- GRU gates + LayerNorm (one block per batch row) ----------------
__global__ void __launch_bounds__(256)
k_gru_ln(const float* __restrict__ Gi, const float* __restrict__ Gh,
         float* __restrict__ h, __bf16* __restrict__ hb,
         const float* __restrict__ lng, const float* __restrict__ lnb) {
  const int b = blockIdx.x, tid = threadIdx.x;
  const int lane = tid & 31, wv = tid >> 5;
  const float* gi = Gi + (size_t)b * G3;
  const float* gh = Gh + (size_t)b * G3;
  float* hrow = h + (size_t)b * HH;
  __shared__ float red[8];

  float vals[2];
  float s = 0.f;
#pragma unroll
  for (int j = 0; j < 2; ++j) {
    int c = tid + j * 256;
    float r = sigmoidf_fast(gi[c] + gh[c]);
    float z = sigmoidf_fast(gi[HH + c] + gh[HH + c]);
    float n = tanhf(gi[2 * HH + c] + r * gh[2 * HH + c]);
    float v = (1.0f - z) * n + z * hrow[c];
    vals[j] = v;
    s += v;
  }
  s = wave_sum(s);
  if (lane == 0) red[wv] = s;
  __syncthreads();
  float tot = 0.f;
#pragma unroll
  for (int i = 0; i < 8; ++i) tot += red[i];
  const float mean = tot * (1.0f / HH);

  float q = 0.f;
#pragma unroll
  for (int j = 0; j < 2; ++j) { float d = vals[j] - mean; q += d * d; }
  q = wave_sum(q);
  __syncthreads();
  if (lane == 0) red[wv] = q;
  __syncthreads();
  float qt = 0.f;
#pragma unroll
  for (int i = 0; i < 8; ++i) qt += red[i];
  const float rstd = rsqrtf(qt * (1.0f / HH) + 1e-5f);

#pragma unroll
  for (int j = 0; j < 2; ++j) {
    int c = tid + j * 256;
    float y = (vals[j] - mean) * rstd * lng[c] + lnb[c];
    hrow[c] = y;
    hb[(size_t)b * HH + c] = (__bf16)y;
  }
}

// ---------------- log-softmax, entropy, Gumbel sample, embed gather ----------------
__global__ void __launch_bounds__(256)
k_sample(const float* __restrict__ Logits, const float* __restrict__ Etab,
         float* __restrict__ oseq, float* __restrict__ olp, float* __restrict__ oent,
         __bf16* __restrict__ xb, int t) {
  const int b = blockIdx.x, tid = threadIdx.x;
  const int lane = tid & 31, wv = tid >> 5;
  const float* lg = Logits + (size_t)b * VV;
  __shared__ float rmax[8], rz[8], rw[8], rbst[8];
  __shared__ int   rbi[8];

  // pass 1: row max
  float m = -3.4e38f;
  for (int v = tid; v < VV; v += 256) m = fmaxf(m, lg[v]);
  m = wave_max(m);
  if (lane == 0) rmax[wv] = m;
  __syncthreads();
  float mm = rmax[0];
#pragma unroll
  for (int i = 1; i < 8; ++i) mm = fmaxf(mm, rmax[i]);

  // pass 2: Z, sum(p*s), Gumbel argmax
  float z = 0.f, sw = 0.f, best = -3.4e38f;
  int besti = 0;
  for (int v = tid; v < VV; v += 256) {
    float s = lg[v] - mm;
    float e = __expf(s);
    z += e;
    sw += e * s;
    float u = u01(((unsigned long long)b << 40) ^
                  ((unsigned long long)t << 20) ^ (unsigned long long)v);
    float sc = s - __logf(-__logf(u));
    if (sc > best) { best = sc; besti = v; }
  }
  z  = wave_sum(z);
  sw = wave_sum(sw);
#pragma unroll
  for (int o = 16; o > 0; o >>= 1) {
    float ob = __shfl_down(best, o, 32);
    int   oi = __shfl_down(besti, o, 32);
    if (ob > best) { best = ob; besti = oi; }
  }
  if (lane == 0) { rz[wv] = z; rw[wv] = sw; rbst[wv] = best; rbi[wv] = besti; }
  __syncthreads();
  float zt = 0.f, swt = 0.f, bb = -3.4e38f;
  int tok = 0;
#pragma unroll
  for (int i = 0; i < 8; ++i) {
    zt += rz[i];
    swt += rw[i];
    if (rbst[i] > bb) { bb = rbst[i]; tok = rbi[i]; }
  }
  const float logZ = __logf(zt);
  if (tid == 0) {
    oseq[(size_t)b * TT + t] = (float)tok;
    olp [(size_t)b * TT + t] = (lg[tok] - mm) - logZ;
    oent[(size_t)b * TT + t] = logZ - swt / zt;
  }
  // next input: x = bf16(embed_table[tok])
  const float* er = Etab + (size_t)tok * EE;
  for (int c = tid; c < EE; c += 256)
    xb[(size_t)b * EE + c] = (__bf16)er[c];
}

// ---------------- launch ----------------
extern "C" void kernel_launch(void* const* d_in, const int* in_sizes, int n_in,
                              void* d_out, int out_size, void* d_ws, size_t ws_size,
                              hipStream_t stream) {
  (void)in_sizes; (void)n_in; (void)out_size; (void)ws_size;
  const float* embedding   = (const float*)d_in[0];
  const float* sos         = (const float*)d_in[2];
  const float* W_ih        = (const float*)d_in[3];
  const float* W_hh        = (const float*)d_in[4];
  const float* b_ih        = (const float*)d_in[5];
  const float* b_hh        = (const float*)d_in[6];
  const float* ln_g        = (const float*)d_in[7];
  const float* ln_b        = (const float*)d_in[8];
  const float* W_out       = (const float*)d_in[9];
  const float* b_out       = (const float*)d_in[10];
  const float* embed_table = (const float*)d_in[11];

  float* out_seq = (float*)d_out;
  float* out_lp  = out_seq + (size_t)BB * TT;
  float* out_ent = out_lp  + (size_t)BB * TT;

  char* w = (char*)d_ws;
  __bf16* Wih_bf  = (__bf16*)w; w += (size_t)G3 * EE * 2;   // 1.5 MB
  __bf16* Whh_bf  = (__bf16*)w; w += (size_t)G3 * HH * 2;   // 1.5 MB
  __bf16* Wout_bf = (__bf16*)w; w += (size_t)VV * HH * 2;   // 2 MB
  __bf16* x_bf    = (__bf16*)w; w += (size_t)BB * EE * 2;   // 8 MB
  __bf16* h_bf    = (__bf16*)w; w += (size_t)BB * HH * 2;   // 8 MB
  float*  h       = (float*) w; w += (size_t)BB * HH * 4;   // 16 MB
  float*  Gi      = (float*) w; w += (size_t)BB * G3 * 4;   // 48 MB
  float*  Gh      = (float*) w; w += (size_t)BB * G3 * 4;   // 48 MB
  float*  Logits  = Gi;  // aliases Gi/Gh region (64 MB <= 96 MB), disjoint in time

  const dim3 blk(256);
  // one-time weight conversion + state init (re-done every call: deterministic)
  k_f32_to_bf16<<<dim3((G3 * EE + 255) / 256), blk, 0, stream>>>(W_ih,  Wih_bf,  G3 * EE);
  k_f32_to_bf16<<<dim3((G3 * HH + 255) / 256), blk, 0, stream>>>(W_hh,  Whh_bf,  G3 * HH);
  k_f32_to_bf16<<<dim3((VV * HH + 255) / 256), blk, 0, stream>>>(W_out, Wout_bf, VV * HH);
  k_init_h<<<dim3((BB * HH + 255) / 256), blk, 0, stream>>>(embedding, h, h_bf, BB * HH);
  k_init_x<<<dim3((BB * EE + 255) / 256), blk, 0, stream>>>(sos, x_bf, BB * EE);

  for (int t = 0; t < TT; ++t) {
    // gi = x @ W_ih^T + b_ih        [8192,1536]
    k_gemm_bf16<<<dim3(BB / 128, G3 / 128), blk, 0, stream>>>(x_bf, Wih_bf, b_ih, Gi, G3, EE);
    // gh = h @ W_hh^T + b_hh        [8192,1536]
    k_gemm_bf16<<<dim3(BB / 128, G3 / 128), blk, 0, stream>>>(h_bf, Whh_bf, b_hh, Gh, G3, HH);
    // GRU gates + LayerNorm -> h, h_bf
    k_gru_ln<<<dim3(BB), blk, 0, stream>>>(Gi, Gh, h, h_bf, ln_g, ln_b);
    // logits = h @ W_out^T + b_out  [8192,2048]
    k_gemm_bf16<<<dim3(BB / 128, VV / 128), blk, 0, stream>>>(h_bf, Wout_bf, b_out, Logits, VV, HH);
    // log-softmax + entropy + sample + embed gather -> outputs, next x
    k_sample<<<dim3(BB), blk, 0, stream>>>(Logits, embed_table, out_seq, out_lp, out_ent, x_bf, t);
  }
}
